// MultiHeadAttention_89670327206273
// MI455X (gfx1250) — matrix operand order repaired and checked by reading, not compiled, not run on hardware
//
#include <hip/hip_runtime.h>

// NEZHA relative-position multi-head attention for MI455X (gfx1250, wave32).
// All matrix contractions run on v_wmma_f32_16x16x32_bf16; fp32 inputs are
// rounded to bf16 (hardware v_cvt, RNE) while staging into LDS. Pure-copy
// stages use GLOBAL_LOAD_ASYNC_TO_LDS_B128 (ASYNCcnt) to bypass VGPRs and
// overlap with WMMA compute. pos_emb (256MB) is reused across all 48 (b,h)
// pairs by expressing the per-j pos einsums as 48-row WMMA GEMMs.

#define DEV __device__ __forceinline__

typedef __bf16 bf16;
typedef __attribute__((ext_vector_type(8)))  __bf16 v8bf;
typedef __attribute__((ext_vector_type(16))) __bf16 v16bf;
typedef __attribute__((ext_vector_type(8)))  float  v8f;

constexpr int Bn = 4, Sn = 1024, Hn = 12, DKn = 64;

// ---------- helpers ----------
DEV unsigned short f2bf(float f) {            // hardware cvt (RNE) via __bf16
  union { __bf16 b; unsigned short u; } v;
  v.b = (__bf16)f;
  return v.u;
}
DEV unsigned pack2(float a, float b) {
  return (unsigned)f2bf(a) | ((unsigned)f2bf(b) << 16);
}
DEV v16bf cat8(v8bf lo, v8bf hi) {
  return __builtin_shufflevector(lo, hi, 0,1,2,3,4,5,6,7,8,9,10,11,12,13,14,15);
}
DEV v8bf ld8(const unsigned short* p) { return *reinterpret_cast<const v8bf*>(p); }

// A fragment (16x32 bf16, M x K): lane<16 holds k in {8h..8h+7} U {16+8h..}
DEV v16bf frag_a(const unsigned short* tile, int ld, int m0, int k0, int lane) {
  const unsigned short* p = tile + (m0 + (lane & 15)) * ld + k0 + 8 * (lane >> 4);
  return cat8(ld8(p), ld8(p + 16));
}
// B fragment (32x16 bf16, K x N), LDS stored n-major with k contiguous:
// lane holds k in {8h..8h+7} U {16+8h..}, matching frag_a
DEV v16bf frag_b(const unsigned short* tile, int ld, int n0, int k0, int lane) {
  const unsigned short* p = tile + (n0 + (lane & 15)) * ld + k0 + 8 * (lane >> 4);
  return cat8(ld8(p), ld8(p + 16));
}
DEV v8f wmma_bf(v16bf a, v16bf b, v8f c) {
  return __builtin_amdgcn_wmma_f32_16x16x32_bf16(false, a, false, b, (short)0, c,
                                                 false, false);
}
DEV v8f zero8() { v8f z = {0.f,0.f,0.f,0.f,0.f,0.f,0.f,0.f}; return z; }

// Async 16B copy global -> LDS (per lane). LDS byte offset = low 32 bits of
// the flat pointer (addrspacecast keeps the wave-relative LDS offset there).
DEV void async_copy16(void* lds_ptr, const void* gptr) {
  unsigned lds_off = (unsigned)(uintptr_t)lds_ptr;
  asm volatile("global_load_async_to_lds_b128 %0, %1, off"
               :: "v"(lds_off), "v"(gptr) : "memory");
}
DEV void async_wait0() { asm volatile("s_wait_asynccnt 0x0" ::: "memory"); }

// ---------- kernel 1/7: generic 4096x768x768 GEMM (projections + output) ----
// mode 0: out bf16 [b][h][s][d]   (qw, kw)
// mode 1: out bf16 [b][h][d][s]   (vw transposed)
// mode 2: out fp32 [m][n] * q_mask[m]  (final projection)
__global__ __launch_bounds__(256, 2) void gemm768_bf16(
    const float* __restrict__ X, const float* __restrict__ W,
    const float* __restrict__ bias, void* __restrict__ out,
    int mode, const float* __restrict__ qmask)
{
  constexpr int LDT = 40;                         // bf16 elems; 80B rows (16B mult)
  __shared__ unsigned short As[2][128 * LDT];
  __shared__ unsigned short Bs[2][64 * LDT];
  const int t = threadIdx.x, lane = t & 31, wave = t >> 5;
  const int mw = wave >> 1, nw = wave & 1;        // 4x2 wave grid
  const int m0g = blockIdx.y * 128, n0g = blockIdx.x * 64;

  v8f acc[2][2];
  #pragma unroll
  for (int i = 0; i < 2; ++i)
    #pragma unroll
    for (int j = 0; j < 2; ++j) acc[i][j] = zero8();

  auto stage = [&](int buf, int kk0) {
    #pragma unroll
    for (int i = 0; i < 4; ++i) {                 // A: 128x32 fp32 -> bf16
      int idx = t + 256 * i, row = idx >> 3, c4 = (idx & 7) << 2;
      float4 f = *reinterpret_cast<const float4*>(&X[(m0g + row) * 768 + kk0 + c4]);
      uint2 pk; pk.x = pack2(f.x, f.y); pk.y = pack2(f.z, f.w);
      *reinterpret_cast<uint2*>(&As[buf][row * LDT + c4]) = pk;
    }
    #pragma unroll
    for (int i = 0; i < 2; ++i) {                 // B: 32x64 fp32, transpose to [n][k]
      int idx = t + 256 * i, kr = idx >> 4, c4 = (idx & 15) << 2;
      float4 f = *reinterpret_cast<const float4*>(&W[(kk0 + kr) * 768 + n0g + c4]);
      Bs[buf][(c4 + 0) * LDT + kr] = f2bf(f.x);
      Bs[buf][(c4 + 1) * LDT + kr] = f2bf(f.y);
      Bs[buf][(c4 + 2) * LDT + kr] = f2bf(f.z);
      Bs[buf][(c4 + 3) * LDT + kr] = f2bf(f.w);
    }
  };

  stage(0, 0);
  __syncthreads();
  for (int ks = 0; ks < 24; ++ks) {
    int buf = ks & 1;
    if (ks + 1 < 24) stage(buf ^ 1, (ks + 1) * 32);
    v16bf af[2], bfb[2];
    #pragma unroll
    for (int i = 0; i < 2; ++i) af[i]  = frag_a(As[buf], LDT, mw * 32 + i * 16, 0, lane);
    #pragma unroll
    for (int j = 0; j < 2; ++j) bfb[j] = frag_b(Bs[buf], LDT, nw * 32 + j * 16, 0, lane);
    #pragma unroll
    for (int i = 0; i < 2; ++i)
      #pragma unroll
      for (int j = 0; j < 2; ++j) acc[i][j] = wmma_bf(af[i], bfb[j], acc[i][j]);
    __syncthreads();
  }

  const int nl = lane & 15, hh = lane >> 4;
  #pragma unroll
  for (int i = 0; i < 2; ++i)
    #pragma unroll
    for (int j = 0; j < 2; ++j) {
      int ncol = n0g + nw * 32 + j * 16 + nl;
      float bv = bias[ncol];
      #pragma unroll
      for (int r = 0; r < 8; ++r) {
        int mrow = m0g + mw * 32 + i * 16 + r + 8 * hh;
        float val = acc[i][j][r] + bv;
        if (mode == 2) {
          ((float*)out)[mrow * 768 + ncol] = val * qmask[mrow];
        } else {
          int b = mrow >> 10, s = mrow & 1023, h = ncol >> 6, d = ncol & 63;
          unsigned short bw = f2bf(val);
          if (mode == 0)
            ((unsigned short*)out)[((b * Hn + h) * Sn + s) * DKn + d] = bw;
          else
            ((unsigned short*)out)[((b * Hn + h) * DKn + d) * Sn + s] = bw;
        }
      }
    }
}

// ---------- kernel 2/7: pos scores: att[bh, j, k] = Q_j(48x64) @ pos_j^T ----
__global__ __launch_bounds__(256, 2) void pos_scores_kernel(
    const unsigned short* __restrict__ qw, const float* __restrict__ pos,
    float* __restrict__ att)
{
  constexpr int LDQ = 72, LDP = 72;               // 144B rows
  __shared__ unsigned short Qs[48 * LDQ];
  __shared__ unsigned short Ps[128 * LDP];        // [k_seq][d]
  const int k0 = blockIdx.x * 128, j = blockIdx.y;
  const int t = threadIdx.x, lane = t & 31, wave = t >> 5;
  const int n0 = wave * 16;                       // 8 waves x 16-col n-tiles

  for (int idx = t; idx < 384; idx += 256) {      // Q_j: 48 rows x 64 bf16, async
    int row = idx >> 3, c = (idx & 7) << 3;
    async_copy16(&Qs[row * LDQ + c], &qw[(row * Sn + j) * 64 + c]);
  }
  #pragma unroll
  for (int i = 0; i < 8; ++i) {                   // pos_j slice: 128x64 fp32->bf16
    int idx = t + 256 * i, kr = idx >> 4, c4 = (idx & 15) << 2;
    float4 f = *reinterpret_cast<const float4*>(&pos[(j * Sn + k0 + kr) * 64 + c4]);
    uint2 pk; pk.x = pack2(f.x, f.y); pk.y = pack2(f.z, f.w);
    *reinterpret_cast<uint2*>(&Ps[kr * LDP + c4]) = pk;
  }
  if (k0 + 128 < Sn) __builtin_prefetch(&pos[(j * Sn + k0 + 128) * 64], 0, 1);
  async_wait0();
  __syncthreads();

  v8f acc[3] = {zero8(), zero8(), zero8()};
  #pragma unroll
  for (int kk = 0; kk < 64; kk += 32) {
    v16bf bfb = frag_b(Ps, LDP, n0, kk, lane);
    #pragma unroll
    for (int mt = 0; mt < 3; ++mt) {
      v16bf af = frag_a(Qs, LDQ, mt * 16, kk, lane);
      acc[mt] = wmma_bf(af, bfb, acc[mt]);
    }
  }
  const int nl = lane & 15, hh = lane >> 4;
  #pragma unroll
  for (int mt = 0; mt < 3; ++mt)
    #pragma unroll
    for (int r = 0; r < 8; ++r) {
      int bh = mt * 16 + r + 8 * hh;
      att[(bh * Sn + j) * Sn + k0 + n0 + nl] = acc[mt][r];
    }
}

// ---------- kernel 3/7: content scores, accumulate: att += qw @ kw^T --------
__global__ __launch_bounds__(256, 2) void qk_scores_kernel(
    const unsigned short* __restrict__ qw, const unsigned short* __restrict__ kw,
    float* __restrict__ att)
{
  constexpr int LDT = 72;
  __shared__ unsigned short As[128 * LDT];
  __shared__ unsigned short Bs[64 * LDT];         // [k_seq][d]
  const int k0 = blockIdx.x * 64, j0 = blockIdx.y * 128, bh = blockIdx.z;
  const int t = threadIdx.x, lane = t & 31, wave = t >> 5;
  const int mw = wave >> 1, nw = wave & 1;
  const unsigned short* qb = qw + bh * Sn * 64;
  const unsigned short* kb = kw + bh * Sn * 64;

  #pragma unroll
  for (int i = 0; i < 4; ++i) {                   // A tile: async copy
    int idx = t + 256 * i, row = idx >> 3, c = (idx & 7) << 3;
    async_copy16(&As[row * LDT + c], &qb[(j0 + row) * 64 + c]);
  }
  #pragma unroll
  for (int i = 0; i < 2; ++i) {                   // B tile: async copy
    int idx = t + 256 * i, row = idx >> 3, c = (idx & 7) << 3;
    async_copy16(&Bs[row * LDT + c], &kb[(k0 + row) * 64 + c]);
  }
  async_wait0();
  __syncthreads();

  const int nl = lane & 15, hh = lane >> 4;
  v8f acc[2][2];
  #pragma unroll
  for (int i = 0; i < 2; ++i)                     // C init from pos scores
    #pragma unroll
    for (int jj = 0; jj < 2; ++jj)
      #pragma unroll
      for (int r = 0; r < 8; ++r)
        acc[i][jj][r] = att[(bh * Sn + j0 + mw * 32 + i * 16 + r + 8 * hh) * Sn +
                            k0 + nw * 32 + jj * 16 + nl];

  #pragma unroll
  for (int kk = 0; kk < 64; kk += 32) {
    v16bf af[2], bfb[2];
    #pragma unroll
    for (int i = 0; i < 2; ++i)  af[i]  = frag_a(As, LDT, mw * 32 + i * 16, kk, lane);
    #pragma unroll
    for (int jj = 0; jj < 2; ++jj) bfb[jj] = frag_b(Bs, LDT, nw * 32 + jj * 16, kk, lane);
    #pragma unroll
    for (int i = 0; i < 2; ++i)
      #pragma unroll
      for (int jj = 0; jj < 2; ++jj) acc[i][jj] = wmma_bf(af[i], bfb[jj], acc[i][jj]);
  }
  #pragma unroll
  for (int i = 0; i < 2; ++i)
    #pragma unroll
    for (int jj = 0; jj < 2; ++jj)
      #pragma unroll
      for (int r = 0; r < 8; ++r)
        att[(bh * Sn + j0 + mw * 32 + i * 16 + r + 8 * hh) * Sn +
            k0 + nw * 32 + jj * 16 + nl] = acc[i][jj][r];
}

// ---------- kernel 4/7: row softmax with key mask ---------------------------
__global__ __launch_bounds__(256) void softmax_kernel(
    float* __restrict__ att, const float* __restrict__ vmask)
{
  const int row = blockIdx.x;                     // bh*S + j
  const int b = row / (Hn * Sn);
  float* r = att + (size_t)row * Sn;
  const float* vmb = vmask + b * Sn;
  const int t = threadIdx.x;
  __shared__ float red[8];

  float x[4], mx = -3.0e38f;
  #pragma unroll
  for (int i = 0; i < 4; ++i) {
    int k = t + (i << 8);
    float v = r[k] * 0.125f - (1.0f - vmb[k]) * 1.0e12f;   // /sqrt(64), mask
    x[i] = v; mx = fmaxf(mx, v);
  }
  #pragma unroll
  for (int off = 16; off; off >>= 1) mx = fmaxf(mx, __shfl_xor(mx, off));
  if ((t & 31) == 0) red[t >> 5] = mx;
  __syncthreads();
  mx = red[0];
  #pragma unroll
  for (int i = 1; i < 8; ++i) mx = fmaxf(mx, red[i]);
  __syncthreads();

  float s = 0.f;
  #pragma unroll
  for (int i = 0; i < 4; ++i) { x[i] = __expf(x[i] - mx); s += x[i]; }
  #pragma unroll
  for (int off = 16; off; off >>= 1) s += __shfl_xor(s, off);
  if ((t & 31) == 0) red[t >> 5] = s;
  __syncthreads();
  s = red[0];
  #pragma unroll
  for (int i = 1; i < 8; ++i) s += red[i];
  float inv = 1.0f / s;
  #pragma unroll
  for (int i = 0; i < 4; ++i) r[t + (i << 8)] = x[i] * inv;
}

// ---------- kernel 5/7: ctx = att @ vw  (per b,h; 1024x64x1024) -------------
__global__ __launch_bounds__(256, 2) void ctx_content_kernel(
    const float* __restrict__ att, const unsigned short* __restrict__ vwT,
    float* __restrict__ ctx)
{
  constexpr int LDT = 40;
  __shared__ unsigned short As[2][128 * LDT];
  __shared__ unsigned short Bs[2][64 * LDT];      // [d][k_seq] direct from vwT
  const int j0 = blockIdx.x * 128, bh = blockIdx.y;
  const int t = threadIdx.x, lane = t & 31, wave = t >> 5;
  const int mw = wave >> 1, nw = wave & 1;
  const float* arow = att + (size_t)(bh * Sn + j0) * Sn;
  const unsigned short* vb = vwT + bh * 64 * Sn;

  v8f acc[2][2];
  #pragma unroll
  for (int i = 0; i < 2; ++i)
    #pragma unroll
    for (int j = 0; j < 2; ++j) acc[i][j] = zero8();

  auto stage = [&](int buf, int kk0) {
    #pragma unroll
    for (int i = 0; i < 4; ++i) {                 // att 128x32 fp32 -> bf16
      int idx = t + 256 * i, row = idx >> 3, c4 = (idx & 7) << 2;
      float4 f = *reinterpret_cast<const float4*>(&arow[row * Sn + kk0 + c4]);
      uint2 pk; pk.x = pack2(f.x, f.y); pk.y = pack2(f.z, f.w);
      *reinterpret_cast<uint2*>(&As[buf][row * LDT + c4]) = pk;
    }
    {                                             // vwT 64x32 bf16: async copy
      int row = t >> 2, c = (t & 3) << 3;
      async_copy16(&Bs[buf][row * LDT + c], &vb[row * Sn + kk0 + c]);
    }
  };

  stage(0, 0);
  async_wait0();
  __syncthreads();
  for (int ks = 0; ks < 32; ++ks) {
    int buf = ks & 1;
    if (ks + 1 < 32) stage(buf ^ 1, (ks + 1) * 32);   // async overlaps compute
    v16bf af[2], bfb[2];
    #pragma unroll
    for (int i = 0; i < 2; ++i)  af[i]  = frag_a(As[buf], LDT, mw * 32 + i * 16, 0, lane);
    #pragma unroll
    for (int j = 0; j < 2; ++j)  bfb[j] = frag_b(Bs[buf], LDT, nw * 32 + j * 16, 0, lane);
    #pragma unroll
    for (int i = 0; i < 2; ++i)
      #pragma unroll
      for (int j = 0; j < 2; ++j) acc[i][j] = wmma_bf(af[i], bfb[j], acc[i][j]);
    async_wait0();
    __syncthreads();
  }

  const int nl = lane & 15, hh = lane >> 4;
  const int b = bh / Hn, h = bh % Hn;
  #pragma unroll
  for (int i = 0; i < 2; ++i)
    #pragma unroll
    for (int j = 0; j < 2; ++j)
      #pragma unroll
      for (int r = 0; r < 8; ++r)
        ctx[(b * Sn + j0 + mw * 32 + i * 16 + r + 8 * hh) * 768 +
            h * 64 + nw * 32 + j * 16 + nl] = acc[i][j][r];
}

// ---------- kernel 6/7: ctx += Att_j(48x1024) @ pos_j(1024x64) --------------
__global__ __launch_bounds__(128, 2) void ctx_pos_kernel(
    const float* __restrict__ att, const float* __restrict__ pos,
    float* __restrict__ ctx)
{
  constexpr int LDA = 72, LDP = 72;
  __shared__ unsigned short As[48 * LDA];         // [bh][k_seq]
  __shared__ unsigned short Ps[64 * LDP];         // [d][k_seq] (transposed stage)
  const int j = blockIdx.x;
  const int t = threadIdx.x, lane = t & 31, wave = t >> 5;
  const int n0 = wave * 16;                       // 4 waves x 16 d-cols
  const int nl = lane & 15, hh = lane >> 4;

  v8f acc[3];
  #pragma unroll
  for (int mt = 0; mt < 3; ++mt)                  // C init from ctx_content
    #pragma unroll
    for (int r = 0; r < 8; ++r) {
      int bh = mt * 16 + r + 8 * hh;
      acc[mt][r] = ctx[((bh / Hn) * Sn + j) * 768 + (bh % Hn) * 64 + n0 + nl];
    }

  for (int kk0 = 0; kk0 < Sn; kk0 += 64) {
    #pragma unroll
    for (int i = 0; i < 6; ++i) {                 // att 48x64 fp32 -> bf16
      int idx = t + 128 * i, row = idx >> 4, c4 = (idx & 15) << 2;
      float4 f = *reinterpret_cast<const float4*>(
          &att[(size_t)(row * Sn + j) * Sn + kk0 + c4]);
      uint2 pk; pk.x = pack2(f.x, f.y); pk.y = pack2(f.z, f.w);
      *reinterpret_cast<uint2*>(&As[row * LDA + c4]) = pk;
    }
    #pragma unroll
    for (int i = 0; i < 8; ++i) {                 // pos 64x64 fp32 -> bf16, transposed
      int idx = t + 128 * i, kr = idx >> 4, c4 = (idx & 15) << 2;
      float4 f = *reinterpret_cast<const float4*>(&pos[(j * Sn + kk0 + kr) * 64 + c4]);
      Ps[(c4 + 0) * LDP + kr] = f2bf(f.x);
      Ps[(c4 + 1) * LDP + kr] = f2bf(f.y);
      Ps[(c4 + 2) * LDP + kr] = f2bf(f.z);
      Ps[(c4 + 3) * LDP + kr] = f2bf(f.w);
    }
    if (kk0 + 64 < Sn) __builtin_prefetch(&pos[(j * Sn + kk0 + 64) * 64], 0, 1);
    __syncthreads();
    #pragma unroll
    for (int kk = 0; kk < 64; kk += 32) {
      v16bf bfb = frag_b(Ps, LDP, n0, kk, lane);
      #pragma unroll
      for (int mt = 0; mt < 3; ++mt) {
        v16bf af = frag_a(As, LDA, mt * 16, kk, lane);
        acc[mt] = wmma_bf(af, bfb, acc[mt]);
      }
    }
    __syncthreads();
  }
  #pragma unroll
  for (int mt = 0; mt < 3; ++mt)
    #pragma unroll
    for (int r = 0; r < 8; ++r) {
      int bh = mt * 16 + r + 8 * hh;
      ctx[((bh / Hn) * Sn + j) * 768 + (bh % Hn) * 64 + n0 + nl] = acc[mt][r];
    }
}

// ---------- launcher --------------------------------------------------------
extern "C" void kernel_launch(void* const* d_in, const int* in_sizes, int n_in,
                              void* d_out, int out_size, void* d_ws, size_t ws_size,
                              hipStream_t stream)
{
  (void)in_sizes; (void)n_in; (void)out_size; (void)ws_size;
  const float* q   = (const float*)d_in[0];
  const float* k   = (const float*)d_in[1];
  const float* v   = (const float*)d_in[2];
  const float* pos = (const float*)d_in[3];
  const float* qm  = (const float*)d_in[4];
  const float* vm  = (const float*)d_in[5];
  const float* Wq  = (const float*)d_in[6];
  const float* bq  = (const float*)d_in[7];
  const float* Wk  = (const float*)d_in[8];
  const float* bk  = (const float*)d_in[9];
  const float* Wv  = (const float*)d_in[10];
  const float* bv  = (const float*)d_in[11];
  const float* Wo  = (const float*)d_in[12];
  const float* bo  = (const float*)d_in[13];

  // workspace layout (bytes):
  //   qw  bf16 [b][h][s][d] :        0 ..  6291456
  //   kw  bf16 [b][h][s][d] :  6291456 .. 12582912
  //   vwT bf16 [b][h][d][s] : 12582912 .. 18874368
  //   ctx fp32 [b][s][h*d]  : 18874368 .. 31457280
  //   att fp32 [bh][j][k]   : 31457280 .. 232783872
  char* ws = (char*)d_ws;
  unsigned short* qw  = (unsigned short*)(ws);
  unsigned short* kw  = (unsigned short*)(ws + 6291456);
  unsigned short* vwT = (unsigned short*)(ws + 12582912);
  float* ctx = (float*)(ws + 18874368);
  float* att = (float*)(ws + 31457280);

  dim3 gg(12, 32);  // N/64 x M/128
  gemm768_bf16<<<gg, 256, 0, stream>>>(q, Wq, bq, qw, 0, nullptr);
  gemm768_bf16<<<gg, 256, 0, stream>>>(k, Wk, bk, kw, 0, nullptr);
  gemm768_bf16<<<gg, 256, 0, stream>>>(v, Wv, bv, vwT, 1, nullptr);

  pos_scores_kernel<<<dim3(8, 1024), 256, 0, stream>>>(qw, pos, att);
  qk_scores_kernel<<<dim3(16, 8, 48), 256, 0, stream>>>(qw, kw, att);
  softmax_kernel<<<dim3(48 * 1024), 256, 0, stream>>>(att, vm);
  ctx_content_kernel<<<dim3(8, 48), 256, 0, stream>>>(att, vwT, ctx);
  ctx_pos_kernel<<<dim3(1024), 128, 0, stream>>>(att, pos, ctx);

  gemm768_bf16<<<gg, 256, 0, stream>>>(ctx, Wo, bo, (float*)d_out, 2, qm);
}
